// LSTM_31748398252646
// MI455X (gfx1250) — matrix operand order repaired and checked by reading, not compiled, not run on hardware
//
#include <hip/hip_runtime.h>

typedef __attribute__((ext_vector_type(16))) _Float16 v16h;
typedef __attribute__((ext_vector_type(8)))  _Float16 v8h;
typedef __attribute__((ext_vector_type(8)))  float    v8f;

#define T_STEPS 336
#define BATCH   4096
#define HDIM    50
#define NREG_F16 98304            // 6 regions * 16 Mtiles * 16 rows * 64 K
#define LDS_BYTES (NREG_F16 * 2)  // 192 KB

union F16x16 { v16h v; v8h h[2]; };
union H8     { v8h h; unsigned u[4]; };

__device__ __forceinline__ v8f zero8f() {
  v8f r;
#pragma unroll
  for (int i = 0; i < 8; ++i) r[i] = 0.0f;
  return r;
}
__device__ __forceinline__ v16h zero16h() {
  v16h r;
#pragma unroll
  for (int i = 0; i < 16; ++i) r[i] = (_Float16)0.0f;
  return r;
}

__device__ __forceinline__ float fsig(float x) {
  // sigmoid via native exp2 + rcp (TRANS ops co-execute with WMMA)
  return __builtin_amdgcn_rcpf(1.0f + __builtin_amdgcn_exp2f(-1.4426950408889634f * x));
}
__device__ __forceinline__ float ftanh(float x) {
  return 2.0f * __builtin_amdgcn_rcpf(1.0f + __builtin_amdgcn_exp2f(-2.8853900817779268f * x)) - 1.0f;
}

__device__ __forceinline__ v8f wmma_f16(v16h a, v16h b, v8f c) {
  return __builtin_amdgcn_wmma_f32_16x16x32_f16(false, a, false, b, (short)0, c, false, false);
}

// A-fragment load from LDS row base p (already offset by chunk*32 + hi*16):
// elements j0..7 at p, j8..15 at p+8 (K = 16 hi + j, matches B)
__device__ __forceinline__ v16h load_afrag(const _Float16* p) {
  F16x16 r;
  r.h[0] = *(const v8h*)(p);
  r.h[1] = *(const v8h*)(p + 8);
  return r.v;
}

// One LSTM layer, one timestep, for a 16-row batch tile held by this wave.
// z^T = W_ih @ in^T + W_hh @ h^T  (A = weights from LDS, B = activations)
// Gate order (PyTorch): i,f,g,o; each gate padded to 64 rows (4 Mtiles).
template<int NIN>
__device__ __forceinline__ void layer_step(const _Float16* WL, int l,
    v16h in0, v16h in1, v8f* cst, v16h& hc0, v16h& hc1, int hi, int m)
{
  const _Float16* ihB = WL + (2 * l)     * 16384 + m * 64 + hi * 16;
  const _Float16* hhB = WL + (2 * l + 1) * 16384 + m * 64 + hi * 16;
  v8h ht[4];
#pragma unroll
  for (int s = 0; s < 4; ++s) {
    v8f z[4];
#pragma unroll
    for (int g = 0; g < 4; ++g) {
      const int mt = g * 4 + s;
      v8f acc = zero8f();
      const _Float16* pa = ihB + mt * 1024;
      acc = wmma_f16(load_afrag(pa), in0, acc);
      if (NIN == 2) acc = wmma_f16(load_afrag(pa + 32), in1, acc);
      const _Float16* ph = hhB + mt * 1024;
      acc = wmma_f16(load_afrag(ph), hc0, acc);
      acc = wmma_f16(load_afrag(ph + 32), hc1, acc);
      z[g] = acc;
    }
#pragma unroll
    for (int r = 0; r < 8; ++r) {
      float ig = fsig(z[0][r]);
      float fg = fsig(z[1][r]);
      float gg = ftanh(z[2][r]);
      float og = fsig(z[3][r]);
      float cn = fg * cst[s][r] + ig * gg;
      cst[s][r] = cn;
      ht[s][r] = (_Float16)(og * ftanh(cn));
    }
  }
  // Repack D-layout h (row H = 16 s + r + 8 hi) into next-step B-fragments
  // (element j of chunk c holds K = 32 c + 16 hi + j): half-wave exchange.
  F16x16 nh[2];
#pragma unroll
  for (int c = 0; c < 2; ++c) {
    H8 keep, send, recv;
    keep.h = ht[2 * c + hi];
    send.h = ht[2 * c + (hi ^ 1)];
#pragma unroll
    for (int q = 0; q < 4; ++q) recv.u[q] = __shfl_xor(send.u[q], 16, 32);
    nh[c].h[0] = hi ? recv.h : keep.h;
    nh[c].h[1] = hi ? keep.h : recv.h;
  }
  if (hi) nh[1].h[0][2] = (_Float16)1.0f;  // constant 1 at K=50 -> bias column
  hc0 = nh[0].v;
  hc1 = nh[1].v;
}

__global__ __launch_bounds__(128, 1) void lstm_fused(
    const float* __restrict__ x, const _Float16* __restrict__ wpre,
    const float* __restrict__ wlin, const float* __restrict__ blin,
    float* __restrict__ out)
{
  extern __shared__ _Float16 WL[];
  const int tid = threadIdx.x;

  // Async bulk copy of pre-padded f16 weight image (192 KB) into LDS.
  for (int i = tid; i < NREG_F16 / 8; i += 128) {
    const uint4* g = (const uint4*)wpre + i;
    unsigned lds = (unsigned)(size_t)WL + (unsigned)i * 16u;
    asm volatile("global_load_async_to_lds_b128 %0, %1, off"
                 :: "v"(lds), "v"(g) : "memory");
  }
  asm volatile("s_wait_asynccnt 0" ::: "memory");
  __syncthreads();

  const int lane = tid & 31;
  const int wave = tid >> 5;
  const int hi   = lane >> 4;
  const int m    = lane & 15;          // batch column within tile / A row
  const int bt   = blockIdx.x * 4 + wave;
  const float* xrow = x + (size_t)(bt * 16 + m) * T_STEPS * 12;

  v8f c0[4], c1[4], c2[4];
#pragma unroll
  for (int s = 0; s < 4; ++s) { c0[s] = zero8f(); c1[s] = zero8f(); c2[s] = zero8f(); }

  F16x16 h1i; h1i.v = zero16h();
  if (hi) h1i.h[0][2] = (_Float16)1.0f;   // K=50 constant present from t=0
  v16h h0c0 = zero16h(), h1c0 = zero16h(), h2c0 = zero16h();
  v16h h0c1 = h1i.v,     h1c1 = h1i.v,    h2c1 = h1i.v;

  for (int t = 0; t < T_STEPS; ++t) {
    // Compiler barrier: keep the (loop-invariant) LDS weight-fragment loads
    // INSIDE the loop as ds_load_b128 instead of hoisting them into ~1400
    // VGPRs worth of values that spill to scratch.
    asm volatile("" ::: "memory");

    // x^T B-fragment: lanes hi==0 hold K=F=j (0..11); everything else zero.
    F16x16 xf; xf.v = zero16h();
    if (!hi) {
      const float4 a = *(const float4*)(xrow + t * 12);
      const float4 b = *(const float4*)(xrow + t * 12 + 4);
      const float4 c = *(const float4*)(xrow + t * 12 + 8);
      xf.h[0][0] = (_Float16)a.x; xf.h[0][1] = (_Float16)a.y;
      xf.h[0][2] = (_Float16)a.z; xf.h[0][3] = (_Float16)a.w;
      xf.h[0][4] = (_Float16)b.x; xf.h[0][5] = (_Float16)b.y;
      xf.h[0][6] = (_Float16)b.z; xf.h[0][7] = (_Float16)b.w;
      xf.h[1][0] = (_Float16)c.x; xf.h[1][1] = (_Float16)c.y;
      xf.h[1][2] = (_Float16)c.z; xf.h[1][3] = (_Float16)c.w;
    }
    layer_step<1>(WL, 0, xf.v, xf.v,  c0, h0c0, h0c1, hi, m);
    layer_step<2>(WL, 1, h0c0, h0c1,  c1, h1c0, h1c1, hi, m);
    layer_step<2>(WL, 2, h1c0, h1c1,  c2, h2c0, h2c1, hi, m);
  }

  // Final linear: out[b] = sum_H wlin[H] * h2[H, b] + blin[0]
  float partial = 0.0f;
  F16x16 f0; f0.v = h2c0;
  F16x16 f1; f1.v = h2c1;
#pragma unroll
  for (int c = 0; c < 2; ++c) {
#pragma unroll
    for (int j = 0; j < 16; ++j) {
      const int H = 32 * c + 16 * hi + j;
      if (H < HDIM) {
        float hv = (float)((c ? f1 : f0).h[j >> 3][j & 7]);
        partial += wlin[H] * hv;
      }
    }
  }
  partial += __shfl_xor(partial, 16, 32);
  if (!hi) out[bt * 16 + m] = partial + blin[0];
}

// ---------------- weight pre-pad kernel ----------------
struct WPtrs {
  const float* wih[3]; const float* whh[3];
  const float* bih[3]; const float* bhh[3];
};

// Region layout (f16): [r=2l+isHH][Mtile(16)][row m(16)][K(64)]
// padded row P = Mtile*16+m = gate*64 + idx; idx<50 valid.
// hh: K<50 = W_hh, K==50 = b_ih+b_hh (bias column), else 0.
// ih: K<in = W_ih, else 0.
__global__ __launch_bounds__(256) void prep_weights(WPtrs p, _Float16* wpre) {
  int idx = blockIdx.x * 256 + threadIdx.x;
  if (idx >= NREG_F16) return;
  int r    = idx >> 14;
  int rem  = idx & 16383;
  int P    = rem >> 6;
  int k    = rem & 63;
  int l    = r >> 1;
  int isHH = r & 1;
  int gate = P >> 6;
  int gidx = P & 63;
  float v = 0.0f;
  if (gidx < HDIM) {
    int row = gate * HDIM + gidx;
    if (isHH) {
      if (k < HDIM)       v = p.whh[l][row * HDIM + k];
      else if (k == HDIM) v = p.bih[l][row] + p.bhh[l][row];
    } else {
      int in = (l == 0) ? 12 : HDIM;
      if (k < in) v = p.wih[l][row * in + k];
    }
  }
  wpre[idx] = (_Float16)v;
}

extern "C" void kernel_launch(void* const* d_in, const int* in_sizes, int n_in,
                              void* d_out, int out_size, void* d_ws, size_t ws_size,
                              hipStream_t stream) {
  (void)in_sizes; (void)n_in; (void)out_size; (void)ws_size;
  const float* x = (const float*)d_in[0];
  WPtrs p;
  for (int l = 0; l < 3; ++l) {
    p.wih[l] = (const float*)d_in[1 + 4 * l];
    p.whh[l] = (const float*)d_in[2 + 4 * l];
    p.bih[l] = (const float*)d_in[3 + 4 * l];
    p.bhh[l] = (const float*)d_in[4 + 4 * l];
  }
  const float* wlin = (const float*)d_in[13];
  const float* blin = (const float*)d_in[14];
  _Float16* wpre = (_Float16*)d_ws;

  prep_weights<<<(NREG_F16 + 255) / 256, 256, 0, stream>>>(p, wpre);

  hipFuncSetAttribute((const void*)lstm_fused,
                      hipFuncAttributeMaxDynamicSharedMemorySize, LDS_BYTES);
  lstm_fused<<<BATCH / 64, 128, LDS_BYTES, stream>>>(x, wpre, wlin, blin,
                                                     (float*)d_out);
}
